// Allegro_54391465837338
// MI455X (gfx1250) — matrix operand order repaired and checked by reading, not compiled, not run on hardware
//
#include <hip/hip_runtime.h>

// ---------------------------------------------------------------------------
// Types for CDNA5 WMMA (wave32, 16x16x32 bf16 -> f32)
// ---------------------------------------------------------------------------
typedef __bf16 v16bf __attribute__((ext_vector_type(16)));
typedef float v8f __attribute__((ext_vector_type(8)));
typedef unsigned short ushort8 __attribute__((ext_vector_type(8)));

union Frag16 {
  v16bf bf;
  ushort8 us[2];
};

__device__ __forceinline__ unsigned short f32_to_bf16(float f) {
  unsigned int u = __float_as_uint(f);
  u += 0x7FFFu + ((u >> 16) & 1u);  // round-to-nearest-even
  return (unsigned short)(u >> 16);
}

// ---------------------------------------------------------------------------
// Problem constants
// ---------------------------------------------------------------------------
#define E_TOT 320000
#define NN 10000

constexpr float SQRT3_F = 1.7320508075688772f;
constexpr float INV_SQRT3_F = 0.57735026918962576f;
constexpr float INV_SQRT2_F = 0.70710678118654752f;
constexpr float INV_SQRT_NEIGH = 0.17677669529663687f;  // 1/sqrt(32)
constexpr float C_OLD = 0.89442719099991588f;           // 1/sqrt(1.25)
constexpr float C_NEW = 0.44721359549995794f;           // 0.5*C_OLD
constexpr float SC40 = 0.15811388300841897f;            // 1/sqrt(40)
constexpr float SC64 = 0.125f;
constexpr float SC96 = 0.10206207261596575f;
constexpr float SC128 = 0.08838834764831845f;
constexpr float SC192 = 0.07216878364870323f;
constexpr float PI_F = 3.14159265358979323846f;
constexpr float BESSEL_C = 0.63245553203367587f;  // sqrt(2/R_MAX)

// ---------------------------------------------------------------------------
// GEMM helpers.  Activations live in LDS as bf16 with wave-private 16-row
// ownership; weights are in global memory pre-transposed (N rows x Kpad cols,
// bf16) so a B-fragment is 32 contiguous bytes per lane.
//
// A 16x32 bf16 frag:  lane l<16  -> row rowBase+l, K kb..kb+15 contiguous
//                     lane l>=16 -> same row,      K kb+16..kb+31 contiguous
// B 32x16 bf16 frag:  lane l<16  -> col nb+l,  K kb..kb+15 contiguous
//                     lane l>=16 -> col nb+l-16, K kb+16..kb+31 contiguous
// C/D f32 16x16:      lane l<16  -> col nb+l, rows rowBase+0..7 (v8f idx v)
//                     lane l>=16 -> col nb+l-16, rows rowBase+8..15
// ---------------------------------------------------------------------------
template <int KCH, int NT>
__device__ __forceinline__ void gemm_acc(const unsigned short* sA, int strideA,
                                         int rowBase,
                                         const unsigned short* __restrict__ gW,
                                         int Kpad, int bk0, v8f* acc) {
  const int l = threadIdx.x & 31;
  const int arow = rowBase + (l & 15);
  const int khalf = (l >> 4) << 4;  // 0 or 16
  v16bf afrag[KCH];
#pragma unroll
  for (int kc = 0; kc < KCH; ++kc) {
    const unsigned short* p = sA + arow * strideA + kc * 32 + khalf;
    Frag16 f;
    f.us[0] = *(const ushort8*)(p);
    f.us[1] = *(const ushort8*)(p + 8);
    afrag[kc] = f.bf;
  }
  const int bcol = l & 15;
  const int bk = (l >> 4) << 4;  // 0 or 16
#pragma unroll
  for (int nt = 0; nt < NT; ++nt) {
#pragma unroll
    for (int kc = 0; kc < KCH; ++kc) {
      const unsigned short* q =
          gW + (size_t)(nt * 16 + bcol) * Kpad + bk0 + kc * 32 + bk;
      Frag16 g;
      g.us[0] = *(const ushort8*)(q);
      g.us[1] = *(const ushort8*)(q + 8);
      acc[nt] = __builtin_amdgcn_wmma_f32_16x16x32_bf16(
          false, afrag[kc], false, g.bf, (short)0, acc[nt], false, false);
    }
  }
}

template <int NT>
__device__ __forceinline__ void zero_acc(v8f* a) {
#pragma unroll
  for (int i = 0; i < NT; ++i)
#pragma unroll
    for (int j = 0; j < 8; ++j) a[i][j] = 0.0f;
}

template <int NT, bool SILU>
__device__ __forceinline__ void store_act(unsigned short* sOut, int strideO,
                                          int rowBase, const v8f* acc,
                                          float scale) {
  const int l = threadIdx.x & 31;
  const int col0 = l & 15;
  const int r0 = rowBase + ((l >> 4) << 3);
#pragma unroll
  for (int nt = 0; nt < NT; ++nt) {
#pragma unroll
    for (int v = 0; v < 8; ++v) {
      float x = acc[nt][v] * scale;
      if (SILU) x = x / (1.0f + __expf(-x));
      sOut[(r0 + v) * strideO + nt * 16 + col0] = f32_to_bf16(x);
    }
  }
}

// ---------------------------------------------------------------------------
// Setup kernels: zero accumulators; convert weights f32 (KxN) -> bf16
// transposed/padded (N x Kpad, zero-fill K padding).
// ---------------------------------------------------------------------------
__global__ void zero_f32(float* __restrict__ p, int n) {
  int i = blockIdx.x * blockDim.x + threadIdx.x;
  const int stride = gridDim.x * blockDim.x;
  for (; i < n; i += stride) p[i] = 0.0f;
}

__global__ void convert_wT(const float* __restrict__ src,
                           unsigned short* __restrict__ dst, int K, int N,
                           int Kpad) {
  const int i = blockIdx.x * blockDim.x + threadIdx.x;
  if (i >= N * Kpad) return;
  const int n = i / Kpad;
  const int k = i - n * Kpad;
  const float v = (k < K) ? src[(size_t)k * N + n] : 0.0f;
  dst[i] = f32_to_bf16(v);
}

// ---------------------------------------------------------------------------
// Phase 1: geometry + 2-body MLP + Wenv0; scatter env0; store lat/w_edge/Y.
// ---------------------------------------------------------------------------
__global__ __launch_bounds__(256) void k_edge_phase1(
    const float* __restrict__ coord, const float* __restrict__ attrs,
    const int* __restrict__ eidx, const unsigned short* __restrict__ w2b0T,
    const unsigned short* __restrict__ w2b1T,
    const unsigned short* __restrict__ w2b2T,
    const unsigned short* __restrict__ wenv0T, float* __restrict__ fcutW,
    float* __restrict__ Yw, float* __restrict__ latW,
    float* __restrict__ wedgeW, float* __restrict__ env0) {
  __shared__ __align__(16) unsigned short sAct[128 * 128];
  __shared__ float sY[128 * 4];
  __shared__ float sFcut[128];
  __shared__ int sCtr[128];
  const int tid = threadIdx.x;
  const int e0 = blockIdx.x * 128;

  if (tid < 128) {
    const int e = e0 + tid;
    const int snd = eidx[e];
    const int ctr = eidx[E_TOT + e];
    sCtr[tid] = ctr;
    const float dx = coord[ctr * 3 + 0] - coord[snd * 3 + 0];
    const float dy = coord[ctr * 3 + 1] - coord[snd * 3 + 1];
    const float dz = coord[ctr * 3 + 2] - coord[snd * 3 + 2];
    const float r = sqrtf(dx * dx + dy * dy + dz * dz + 1e-12f);
    const float u = r * 0.2f;
    const float u2 = u * u, u3 = u2 * u, u6 = u3 * u3, u7 = u6 * u, u8 = u7 * u;
    float fc = 1.0f - 28.0f * u6 + 48.0f * u7 - 21.0f * u8;
    if (u >= 1.0f) fc = 0.0f;
    sFcut[tid] = fc;
    fcutW[e] = fc;
    const float invr = 1.0f / r;
    const float y1 = SQRT3_F * dx * invr;
    const float y2 = SQRT3_F * dy * invr;
    const float y3 = SQRT3_F * dz * invr;
    sY[tid * 4 + 0] = 1.0f;
    sY[tid * 4 + 1] = y1;
    sY[tid * 4 + 2] = y2;
    sY[tid * 4 + 3] = y3;
    Yw[(size_t)e * 4 + 0] = 1.0f;
    Yw[(size_t)e * 4 + 1] = y1;
    Yw[(size_t)e * 4 + 2] = y2;
    Yw[(size_t)e * 4 + 3] = y3;
    // lat_in = [attrs[center](16), attrs[sender](16), bessel*fcut(8), pad]
    unsigned short* row = sAct + tid * 128;
    const float* ac = attrs + (size_t)ctr * 16;
    const float* as = attrs + (size_t)snd * 16;
#pragma unroll
    for (int i = 0; i < 16; ++i) {
      row[i] = f32_to_bf16(ac[i]);
      row[16 + i] = f32_to_bf16(as[i]);
    }
#pragma unroll
    for (int n = 1; n <= 8; ++n)
      row[31 + n] = f32_to_bf16(BESSEL_C * __sinf((float)n * PI_F * u) * invr * fc);
#pragma unroll
    for (int i = 40; i < 64; ++i) row[i] = 0;
  }
  __syncthreads();

  const int rb = (tid >> 5) * 16;
  const int l = tid & 31;
  const int col0 = l & 15;
  const int r0 = rb + ((l >> 4) << 3);

  {  // layer 1: [128x64(K=40)] @ W2b0 -> 64, silu
    v8f acc[4];
    zero_acc<4>(acc);
    gemm_acc<2, 4>(sAct, 128, rb, w2b0T, 64, 0, acc);
    store_act<4, true>(sAct, 128, rb, acc, SC40);
  }
  {  // layer 2: 64 -> 128, silu
    v8f acc[8];
    zero_acc<8>(acc);
    gemm_acc<2, 8>(sAct, 128, rb, w2b1T, 64, 0, acc);
    store_act<8, true>(sAct, 128, rb, acc, SC64);
  }
  {  // layer 3: 128 -> 128, *fcut -> lat (LDS bf16 + global f32)
    v8f acc[8];
    zero_acc<8>(acc);
    gemm_acc<4, 8>(sAct, 128, rb, w2b2T, 128, 0, acc);
#pragma unroll
    for (int nt = 0; nt < 8; ++nt)
#pragma unroll
      for (int v = 0; v < 8; ++v) {
        const int row = r0 + v, col = nt * 16 + col0;
        const float x = acc[nt][v] * SC128 * sFcut[row];
        sAct[row * 128 + col] = f32_to_bf16(x);
        latW[(size_t)(e0 + row) * 128 + col] = x;
      }
  }
  {  // w_all = lat @ Wenv0 / sqrt(128); cols<64 -> w_edge, cols>=64 -> env0
    v8f acc[8];
    zero_acc<8>(acc);
    gemm_acc<4, 8>(sAct, 128, rb, wenv0T, 128, 0, acc);
#pragma unroll
    for (int nt = 0; nt < 8; ++nt)
#pragma unroll
      for (int v = 0; v < 8; ++v) {
        const int row = r0 + v, col = nt * 16 + col0;
        const float w = acc[nt][v] * SC128;
        if (col < 64) {
          wedgeW[(size_t)(e0 + row) * 64 + col] = w;
        } else {
          const int j = col - 64;
          const int m = j >> 1;
          float* ep = env0 + ((size_t)sCtr[row] * 32 + m) * 4;
          if ((j & 1) == 0) {
            atomicAdd(ep + 0, w);  // Y0 == 1
          } else {
            atomicAdd(ep + 1, w * sY[row * 4 + 1]);
            atomicAdd(ep + 2, w * sY[row * 4 + 2]);
            atomicAdd(ep + 3, w * sY[row * 4 + 3]);
          }
        }
      }
  }
}

// ---------------------------------------------------------------------------
// Phase 2: tensor products, s_mix/v_mix, latent MLP + residual, Wenv1/env1.
// ---------------------------------------------------------------------------
__global__ __launch_bounds__(256) void k_edge_phase2(
    const int* __restrict__ eidx, const unsigned short* __restrict__ ws0T,
    const unsigned short* __restrict__ wv0T,
    const unsigned short* __restrict__ wlat0T,
    const unsigned short* __restrict__ wlat1T,
    const unsigned short* __restrict__ wenv1T, const float* __restrict__ fcutW,
    const float* __restrict__ Yw, float* __restrict__ latW,
    const float* __restrict__ wedgeW, float* __restrict__ smixW,
    float* __restrict__ vmixW, const float* __restrict__ env0,
    float* __restrict__ env1) {
  __shared__ __align__(16) unsigned short sBig[128 * 128];  // vin / lat stage
  __shared__ __align__(16) unsigned short sScal[128 * 64];  // scal0 stage
  __shared__ float sY[128 * 4];
  __shared__ float sFcut[128];
  __shared__ int sCtr[128];
  const int tid = threadIdx.x;
  const int e0 = blockIdx.x * 128;

  if (tid < 128) {
    const int e = e0 + tid;
    sCtr[tid] = eidx[E_TOT + e];
    sFcut[tid] = fcutW[e];
#pragma unroll
    for (int c = 0; c < 4; ++c) sY[tid * 4 + c] = Yw[(size_t)e * 4 + c];
  }
  __syncthreads();

  const int rb = (tid >> 5) * 16;
  const int l = tid & 31;
  const int col0 = l & 15;
  const int r0 = rb + ((l >> 4) << 3);

  // v_mix component passes (scal0 staged on pass 0)
#pragma unroll
  for (int c = 0; c < 3; ++c) {
    for (int idx = tid; idx < 128 * 32; idx += 256) {
      const int el = idx >> 5;
      const int m = idx & 31;
      const size_t e = (size_t)(e0 + el);
      const float w0 = wedgeW[e * 64 + 2 * m];
      const float w1 = wedgeW[e * 64 + 2 * m + 1];
      const float* ep = env0 + ((size_t)sCtr[el] * 32 + m) * 4;
      const float es = ep[0] * INV_SQRT_NEIGH;
      float ev[3], fv[3];
#pragma unroll
      for (int d = 0; d < 3; ++d) {
        ev[d] = ep[1 + d] * INV_SQRT_NEIGH;
        fv[d] = w1 * sY[el * 4 + 1 + d];
      }
      const float fs = w0;
      if (c == 0) {
        const float p0 = fs * es;
        const float p1 =
            (fv[0] * ev[0] + fv[1] * ev[1] + fv[2] * ev[2]) * INV_SQRT3_F;
        sScal[el * 64 + m] = f32_to_bf16(p0);
        sScal[el * 64 + 32 + m] = f32_to_bf16(p1);
      }
      const int ca = (c + 1) % 3, cb = (c + 2) % 3;
      sBig[el * 96 + m] = f32_to_bf16(fs * ev[c]);                 // p2
      sBig[el * 96 + 32 + m] = f32_to_bf16(fv[c] * es);            // p3
      sBig[el * 96 + 64 + m] =
          f32_to_bf16((fv[ca] * ev[cb] - fv[cb] * ev[ca]) * INV_SQRT2_F);  // p4
    }
    __syncthreads();
    v8f accv[2];
    zero_acc<2>(accv);
    gemm_acc<3, 2>(sBig, 96, rb, wv0T, 96, 0, accv);
#pragma unroll
    for (int nt = 0; nt < 2; ++nt)
#pragma unroll
      for (int v = 0; v < 8; ++v) {
        const int row = r0 + v, col = nt * 16 + col0;
        vmixW[(size_t)(e0 + row) * 96 + c * 32 + col] = accv[nt][v] * SC96;
      }
    __syncthreads();
  }

  {  // s_mix = scal0 @ Ws0 / sqrt(64)
    v8f accs[2];
    zero_acc<2>(accs);
    gemm_acc<2, 2>(sScal, 64, rb, ws0T, 64, 0, accs);
#pragma unroll
    for (int nt = 0; nt < 2; ++nt)
#pragma unroll
      for (int v = 0; v < 8; ++v) {
        const int row = r0 + v, col = nt * 16 + col0;
        smixW[(size_t)(e0 + row) * 32 + col] = accs[nt][v] * SC64;
      }
  }

  // stage lat (bf16) for latent MLP
  for (int idx = tid; idx < 128 * 128; idx += 256) {
    const int el = idx >> 7, col = idx & 127;
    sBig[el * 128 + col] = f32_to_bf16(latW[(size_t)(e0 + el) * 128 + col]);
  }
  __syncthreads();

  v8f acc[8];
  // hidden: concat(lat[128], scal0[64]) @ Wlat0 (K=192) , silu
  zero_acc<8>(acc);
  gemm_acc<4, 8>(sBig, 128, rb, wlat0T, 192, 0, acc);
  gemm_acc<2, 8>(sScal, 64, rb, wlat0T, 192, 128, acc);
  store_act<8, true>(sBig, 128, rb, acc, SC192);
  // out: @ Wlat1, *fcut, residual with c_old/c_new; update latW + LDS
  zero_acc<8>(acc);
  gemm_acc<4, 8>(sBig, 128, rb, wlat1T, 128, 0, acc);
#pragma unroll
  for (int nt = 0; nt < 8; ++nt)
#pragma unroll
    for (int v = 0; v < 8; ++v) {
      const int row = r0 + v, col = nt * 16 + col0;
      const size_t gi = (size_t)(e0 + row) * 128 + col;
      const float nl = acc[nt][v] * SC128 * sFcut[row];
      const float latv = C_OLD * latW[gi] + C_NEW * nl;
      latW[gi] = latv;
      sBig[row * 128 + col] = f32_to_bf16(latv);
    }

  {  // w_env1 = lat @ Wenv1 / sqrt(128) -> env1 scatter
    v8f acc4[4];
    zero_acc<4>(acc4);
    gemm_acc<4, 4>(sBig, 128, rb, wenv1T, 128, 0, acc4);
#pragma unroll
    for (int nt = 0; nt < 4; ++nt)
#pragma unroll
      for (int v = 0; v < 8; ++v) {
        const int row = r0 + v, col = nt * 16 + col0;
        const float w = acc4[nt][v] * SC128;
        const int m = col >> 1;
        float* ep = env1 + ((size_t)sCtr[row] * 32 + m) * 4;
        if ((col & 1) == 0) {
          atomicAdd(ep + 0, w);
        } else {
          atomicAdd(ep + 1, w * sY[row * 4 + 1]);
          atomicAdd(ep + 2, w * sY[row * 4 + 2]);
          atomicAdd(ep + 3, w * sY[row * 4 + 3]);
        }
      }
  }
}

// ---------------------------------------------------------------------------
// Phase 3: scal1 from env1 gather, final MLP + residual, node reduction.
// ---------------------------------------------------------------------------
__global__ __launch_bounds__(256) void k_edge_phase3(
    const int* __restrict__ eidx, const unsigned short* __restrict__ wfin0T,
    const unsigned short* __restrict__ wfin1T, const float* __restrict__ fcutW,
    const float* __restrict__ latW, const float* __restrict__ smixW,
    const float* __restrict__ vmixW, const float* __restrict__ env1,
    float* __restrict__ out) {
  __shared__ __align__(16) unsigned short sBig[128 * 128];
  __shared__ __align__(16) unsigned short sScal[128 * 64];
  __shared__ float sFcut[128];
  __shared__ int sCtr[128];
  const int tid = threadIdx.x;
  const int e0 = blockIdx.x * 128;

  if (tid < 128) {
    const int e = e0 + tid;
    sCtr[tid] = eidx[E_TOT + e];
    sFcut[tid] = fcutW[e];
  }
  __syncthreads();

  // scal1 = [q0 = s_mix*env1_s, q1 = sum_c v_mix*env1_v / sqrt(3)]
  for (int idx = tid; idx < 128 * 32; idx += 256) {
    const int el = idx >> 5, m = idx & 31;
    const size_t e = (size_t)(e0 + el);
    const float* ep = env1 + ((size_t)sCtr[el] * 32 + m) * 4;
    const float q0 = smixW[e * 32 + m] * ep[0] * INV_SQRT_NEIGH;
    const float* vp = vmixW + e * 96;
    const float q1 = (vp[0 * 32 + m] * ep[1] + vp[1 * 32 + m] * ep[2] +
                      vp[2 * 32 + m] * ep[3]) *
                     INV_SQRT_NEIGH * INV_SQRT3_F;
    sScal[el * 64 + m] = f32_to_bf16(q0);
    sScal[el * 64 + 32 + m] = f32_to_bf16(q1);
  }
  // stage lat (bf16)
  for (int idx = tid; idx < 128 * 128; idx += 256) {
    const int el = idx >> 7, col = idx & 127;
    sBig[el * 128 + col] = f32_to_bf16(latW[(size_t)(e0 + el) * 128 + col]);
  }
  __syncthreads();

  const int rb = (tid >> 5) * 16;
  const int l = tid & 31;
  const int col0 = l & 15;
  const int r0 = rb + ((l >> 4) << 3);

  v8f acc[8];
  zero_acc<8>(acc);
  gemm_acc<4, 8>(sBig, 128, rb, wfin0T, 192, 0, acc);
  gemm_acc<2, 8>(sScal, 64, rb, wfin0T, 192, 128, acc);
  store_act<8, true>(sBig, 128, rb, acc, SC192);

  zero_acc<8>(acc);
  gemm_acc<4, 8>(sBig, 128, rb, wfin1T, 128, 0, acc);
#pragma unroll
  for (int nt = 0; nt < 8; ++nt)
#pragma unroll
    for (int v = 0; v < 8; ++v) {
      const int row = r0 + v, col = nt * 16 + col0;
      const float nl = acc[nt][v] * SC128 * sFcut[row];
      const float latv =
          C_OLD * latW[(size_t)(e0 + row) * 128 + col] + C_NEW * nl;
      atomicAdd(out + (size_t)sCtr[row] * 128 + col, latv * INV_SQRT_NEIGH);
    }
}

// ---------------------------------------------------------------------------
// Launch
// ---------------------------------------------------------------------------
extern "C" void kernel_launch(void* const* d_in, const int* in_sizes, int n_in,
                              void* d_out, int out_size, void* d_ws,
                              size_t ws_size, hipStream_t stream) {
  const float* coord = (const float*)d_in[0];
  const float* attrs = (const float*)d_in[1];
  const int* eidx = (const int*)d_in[2];
  const float* W2b0 = (const float*)d_in[3];
  const float* W2b1 = (const float*)d_in[4];
  const float* W2b2 = (const float*)d_in[5];
  const float* Wenv0 = (const float*)d_in[6];
  const float* Wlat0 = (const float*)d_in[7];
  const float* Wlat1 = (const float*)d_in[8];
  const float* Ws0 = (const float*)d_in[9];
  const float* Wv0 = (const float*)d_in[10];
  const float* Wenv1 = (const float*)d_in[11];
  const float* Wfin0 = (const float*)d_in[12];
  const float* Wfin1 = (const float*)d_in[13];
  float* out = (float*)d_out;

  char* ws = (char*)d_ws;
  size_t off = 0;
  auto take = [&](size_t bytes) -> char* {
    off = (off + 255) & ~(size_t)255;
    char* p = ws + off;
    off += bytes;
    return p;
  };
  auto takeW = [&](int n, int kpad) -> unsigned short* {
    return (unsigned short*)take((size_t)n * kpad * sizeof(unsigned short));
  };
  unsigned short* w2b0T = takeW(64, 64);
  unsigned short* w2b1T = takeW(128, 64);
  unsigned short* w2b2T = takeW(128, 128);
  unsigned short* wenv0T = takeW(128, 128);
  unsigned short* wlat0T = takeW(128, 192);
  unsigned short* wlat1T = takeW(128, 128);
  unsigned short* ws0T = takeW(32, 64);
  unsigned short* wv0T = takeW(32, 96);
  unsigned short* wenv1T = takeW(64, 128);
  unsigned short* wfin0T = takeW(128, 192);
  unsigned short* wfin1T = takeW(128, 128);
  float* fcutW = (float*)take((size_t)E_TOT * 4);
  float* Yw = (float*)take((size_t)E_TOT * 4 * 4);
  float* latW = (float*)take((size_t)E_TOT * 128 * 4);
  float* wedgeW = (float*)take((size_t)E_TOT * 64 * 4);
  float* smixW = (float*)take((size_t)E_TOT * 32 * 4);
  float* vmixW = (float*)take((size_t)E_TOT * 96 * 4);
  float* env0 = (float*)take((size_t)NN * 128 * 4);
  float* env1 = (float*)take((size_t)NN * 128 * 4);

  zero_f32<<<1024, 256, 0, stream>>>(env0, NN * 128);
  zero_f32<<<1024, 256, 0, stream>>>(env1, NN * 128);
  zero_f32<<<1024, 256, 0, stream>>>(out, NN * 128);

  auto conv = [&](const float* src, unsigned short* dst, int K, int N,
                  int Kpad) {
    const int total = N * Kpad;
    convert_wT<<<(total + 255) / 256, 256, 0, stream>>>(src, dst, K, N, Kpad);
  };
  conv(W2b0, w2b0T, 40, 64, 64);
  conv(W2b1, w2b1T, 64, 128, 64);
  conv(W2b2, w2b2T, 128, 128, 128);
  conv(Wenv0, wenv0T, 128, 128, 128);
  conv(Wlat0, wlat0T, 192, 128, 192);
  conv(Wlat1, wlat1T, 128, 128, 128);
  conv(Ws0, ws0T, 64, 32, 64);
  conv(Wv0, wv0T, 96, 32, 96);
  conv(Wenv1, wenv1T, 128, 64, 128);
  conv(Wfin0, wfin0T, 192, 128, 192);
  conv(Wfin1, wfin1T, 128, 128, 128);

  const int nblocks = E_TOT / 128;  // 2500
  k_edge_phase1<<<nblocks, 256, 0, stream>>>(coord, attrs, eidx, w2b0T, w2b1T,
                                             w2b2T, wenv0T, fcutW, Yw, latW,
                                             wedgeW, env0);
  k_edge_phase2<<<nblocks, 256, 0, stream>>>(eidx, ws0T, wv0T, wlat0T, wlat1T,
                                             wenv1T, fcutW, Yw, latW, wedgeW,
                                             smixW, vmixW, env0, env1);
  k_edge_phase3<<<nblocks, 256, 0, stream>>>(eidx, wfin0T, wfin1T, fcutW, latW,
                                             smixW, vmixW, env1, out);
  (void)in_sizes;
  (void)n_in;
  (void)out_size;
  (void)ws_size;
}
